// LSTM_29669634081181
// MI455X (gfx1250) — matrix operand order repaired and checked by reading, not compiled, not run on hardware
//
#include <hip/hip_runtime.h>

typedef _Float16 v16h __attribute__((ext_vector_type(16)));
typedef _Float16 h8   __attribute__((ext_vector_type(8)));
typedef float    v8f  __attribute__((ext_vector_type(8)));

#define DEVI __device__ __forceinline__

namespace {
constexpr int BATCH  = 256;
constexpr int TSTEPS = 512;
constexpr int DIN0   = 64;
constexpr int HID    = 128;
constexpr int NOUT   = 40;
constexpr int STR    = 136;   // padded LDS row stride in halfs (avoids bank conflicts)
}

// gfx1250 has a hardware V_TANH_F32 transcendental; use it if the builtin exists.
#if __has_builtin(__builtin_amdgcn_tanhf)
DEVI float fast_tanh(float x) { return __builtin_amdgcn_tanhf(x); }
DEVI float fast_sig(float x)  { return fmaf(0.5f, __builtin_amdgcn_tanhf(0.5f * x), 0.5f); }
#else
DEVI float fast_tanh(float x) { return tanhf(x); }
DEVI float fast_sig(float x)  { return 1.0f / (1.0f + __expf(-x)); }
#endif

// A-operand fragment (16x32 f16, §7.12.2): lane<16 -> e0..7=K0..7, e8..15=K16..23;
// lane>=16 -> e0..7=K8..15, e8..15=K24..31.  rowp = this lane's row (M = lane&15).
DEVI v16h fragA(const _Float16* rowp, int kbase, int hi) {
  h8 lo = *(const h8*)(rowp + kbase + hi * 8);
  h8 hh = *(const h8*)(rowp + kbase + 16 + hi * 8);
  return __builtin_shufflevector(lo, hh, 0,1,2,3,4,5,6,7,8,9,10,11,12,13,14,15);
}

DEVI v8f wmma_f16(v16h a, v16h b, v8f c) {
  // D = A(16x32) * B(32x16) + C, f32 accumulate
  return __builtin_amdgcn_wmma_f32_16x16x32_f16(false, a, false, b, (short)0, c,
                                                false, false);
}

__global__ void cvt_f16_kernel(const float* __restrict__ s,
                               _Float16* __restrict__ d, int n) {
  int i = blockIdx.x * 256 + threadIdx.x;
  if (i < n) d[i] = (_Float16)s[i];
}

__global__ void bias_sum_kernel(const float* __restrict__ a,
                                const float* __restrict__ b,
                                float* __restrict__ o, int n) {
  int i = blockIdx.x * 256 + threadIdx.x;
  if (i < n) o[i] = a[i] + b[i];
}

// Persistent per-layer LSTM kernel: one block = 16 batch rows, 8 waves,
// wave w owns hidden columns [16w, 16w+16) of each gate. Weights register-resident.
template <int DIN, bool WRITEY, bool WRITEH>
__global__ __launch_bounds__(256)
void lstm_layer_kernel(const _Float16* __restrict__ in,    // [B, T, DIN] f16
                       const _Float16* __restrict__ Wi,    // [4H, DIN]  f16 row-major
                       const _Float16* __restrict__ Wh,    // [4H, H]    f16 row-major
                       const float*    __restrict__ bias,  // [4H] (bih+bhh)
                       _Float16* __restrict__ yout,        // [B, T, H] f16 (if WRITEY)
                       float*    __restrict__ hlast)       // [B, H] f32 (if WRITEH)
{
  constexpr int KI = DIN / 32;
  __shared__ _Float16 hbuf[2][16 * STR];   // double-buffered h_{t-1}/h_t (f16)

  const int tid  = threadIdx.x;
  const int wv   = tid >> 5;
  const int lane = tid & 31;
  const int r    = lane & 15;      // row within 16 (batch row / weight row)
  const int hi   = lane >> 4;      // half-wave selector for K split
  const int col  = wv * 16 + r;    // hidden column N owned by this lane
  const int b0   = blockIdx.x * 16;

  for (int i = tid; i < 16 * STR; i += 256) hbuf[0][i] = (_Float16)0.f;

  // Preload weight B-fragments: B(32x16) layout matches A: lanes0-15: K0-7,K16-23; lanes16-31: K8-15,K24-31
  // -> per lane, two 8-half chunks from weight row `col` at kb*32 + hi*8 and kb*32 + 16 + hi*8.
  v16h wiF[4][KI];
  v16h whF[4][4];
  float bg[4];
#pragma unroll
  for (int g = 0; g < 4; ++g) {
    const _Float16* wir = Wi + (size_t)(g * HID + col) * DIN;
#pragma unroll
    for (int kb = 0; kb < KI; ++kb)
      wiF[g][kb] = fragA(wir, kb * 32, hi);
    const _Float16* whr = Wh + (size_t)(g * HID + col) * HID;
#pragma unroll
    for (int kb = 0; kb < 4; ++kb)
      whF[g][kb] = fragA(whr, kb * 32, hi);
    bg[g] = bias[g * HID + col];
  }

  v8f cs;   // cell-state tile [16x16] in C/D layout (M = k + 8*hi, N = col)
#pragma unroll
  for (int k = 0; k < 8; ++k) cs[k] = 0.f;

  __syncthreads();

  const _Float16* xrow = in + (size_t)(b0 + r) * TSTEPS * DIN;  // contiguous in t

#pragma unroll 2
  for (int t = 0; t < TSTEPS; ++t) {
    const int cur = t & 1, nxt = cur ^ 1;

    v8f acc[4];
#pragma unroll
    for (int g = 0; g < 4; ++g)
#pragma unroll
      for (int k = 0; k < 8; ++k) acc[g][k] = bg[g];

    // input contribution: gates += x_t @ Wih^T
    v16h ax[KI];
#pragma unroll
    for (int kb = 0; kb < KI; ++kb) ax[kb] = fragA(xrow, kb * 32, hi);
#pragma unroll
    for (int g = 0; g < 4; ++g)
#pragma unroll
      for (int kb = 0; kb < KI; ++kb)
        acc[g] = wmma_f16(ax[kb], wiF[g][kb], acc[g]);

    // speculative prefetch of next timestep's x row (safe past end-of-buffer)
    __builtin_prefetch(xrow + DIN, 0, 3);

    // recurrent contribution: gates += h_{t-1} @ Whh^T  (h from LDS)
    const _Float16* hrow = &hbuf[cur][r * STR];
    v16h ah[4];
#pragma unroll
    for (int kb = 0; kb < 4; ++kb) ah[kb] = fragA(hrow, kb * 32, hi);
#pragma unroll
    for (int g = 0; g < 4; ++g)
#pragma unroll
      for (int kb = 0; kb < 4; ++kb)
        acc[g] = wmma_f16(ah[kb], whF[g][kb], acc[g]);

    // pointwise LSTM cell update on the D tiles (gate order i, f, g, o)
#pragma unroll
    for (int k = 0; k < 8; ++k) {
      float ig = fast_sig(acc[0][k]);
      float fg = fast_sig(acc[1][k]);
      float gg = fast_tanh(acc[2][k]);
      float og = fast_sig(acc[3][k]);
      float c  = fg * cs[k] + ig * gg;
      cs[k] = c;
      float h = og * fast_tanh(c);
      const int M = k + hi * 8;
      hbuf[nxt][M * STR + col] = (_Float16)h;
      if (WRITEY)
        yout[((size_t)(b0 + M) * TSTEPS + t) * HID + col] = (_Float16)h;
    }
    xrow += DIN;
    __syncthreads();
  }

  // h_{T-1} ends up in hbuf[0] (T even); barrier at loop end makes it visible.
  if (WRITEH) {
#pragma unroll
    for (int k = 0; k < 8; ++k) {
      const int M = k + hi * 8;
      hlast[(b0 + M) * HID + col] = (float)hbuf[0][M * STR + col];
    }
  }
}

__global__ void fc_kernel(const float* __restrict__ h,   // [B, H]
                          const float* __restrict__ W,   // [NOUT, H]
                          const float* __restrict__ b,   // [NOUT]
                          float* __restrict__ out) {     // [B, NOUT]
  int i = blockIdx.x * 256 + threadIdx.x;
  if (i >= BATCH * NOUT) return;
  int bi = i / NOUT, j = i % NOUT;
  const float* hr = h + bi * HID;
  const float* wr = W + j * HID;
  float s = b[j];
#pragma unroll 4
  for (int k = 0; k < HID; ++k) s = fmaf(hr[k], wr[k], s);
  out[i] = s;
}

extern "C" void kernel_launch(void* const* d_in, const int* in_sizes, int n_in,
                              void* d_out, int out_size, void* d_ws, size_t ws_size,
                              hipStream_t stream) {
  (void)in_sizes; (void)n_in; (void)out_size; (void)ws_size;
  const float* x    = (const float*)d_in[0];
  const float* Wih0 = (const float*)d_in[1];
  const float* Whh0 = (const float*)d_in[2];
  const float* bih0 = (const float*)d_in[3];
  const float* bhh0 = (const float*)d_in[4];
  const float* Wih1 = (const float*)d_in[5];
  const float* Whh1 = (const float*)d_in[6];
  const float* bih1 = (const float*)d_in[7];
  const float* bhh1 = (const float*)d_in[8];
  const float* Wfc  = (const float*)d_in[9];
  const float* bfc  = (const float*)d_in[10];

  size_t off = 0;
  char* wsb = (char*)d_ws;
  auto take = [&](size_t bytes) -> void* {
    void* p = wsb + off;
    off += (bytes + 255) & ~(size_t)255;
    return p;
  };
  _Float16* x16  = (_Float16*)take((size_t)BATCH * TSTEPS * DIN0 * 2);
  _Float16* y016 = (_Float16*)take((size_t)BATCH * TSTEPS * HID * 2);
  _Float16* wi0  = (_Float16*)take((size_t)4 * HID * DIN0 * 2);
  _Float16* wh0  = (_Float16*)take((size_t)4 * HID * HID * 2);
  _Float16* wi1  = (_Float16*)take((size_t)4 * HID * HID * 2);
  _Float16* wh1  = (_Float16*)take((size_t)4 * HID * HID * 2);
  float* b0      = (float*)take(4 * HID * 4);
  float* b1      = (float*)take(4 * HID * 4);
  float* h1last  = (float*)take((size_t)BATCH * HID * 4);

  const int nx = BATCH * TSTEPS * DIN0;
  cvt_f16_kernel<<<(nx + 255) / 256, 256, 0, stream>>>(x, x16, nx);
  cvt_f16_kernel<<<(4 * HID * DIN0 + 255) / 256, 256, 0, stream>>>(Wih0, wi0, 4 * HID * DIN0);
  cvt_f16_kernel<<<(4 * HID * HID + 255) / 256, 256, 0, stream>>>(Whh0, wh0, 4 * HID * HID);
  cvt_f16_kernel<<<(4 * HID * HID + 255) / 256, 256, 0, stream>>>(Wih1, wi1, 4 * HID * HID);
  cvt_f16_kernel<<<(4 * HID * HID + 255) / 256, 256, 0, stream>>>(Whh1, wh1, 4 * HID * HID);
  bias_sum_kernel<<<2, 256, 0, stream>>>(bih0, bhh0, b0, 4 * HID);
  bias_sum_kernel<<<2, 256, 0, stream>>>(bih1, bhh1, b1, 4 * HID);

  lstm_layer_kernel<DIN0, true,  false><<<BATCH / 16, 256, 0, stream>>>(x16,  wi0, wh0, b0, y016, nullptr);
  lstm_layer_kernel<HID,  false, true ><<<BATCH / 16, 256, 0, stream>>>(y016, wi1, wh1, b1, nullptr, h1last);

  fc_kernel<<<(BATCH * NOUT + 255) / 256, 256, 0, stream>>>(h1last, Wfc, bfc, (float*)d_out);
}
